// Encoder_87531433492740
// MI455X (gfx1250) — compile-verified, matrix operand order checked by reading, not run on hardware
//
#include <hip/hip_runtime.h>
#include <hip/hip_bf16.h>

// ---------------------------------------------------------------------------
// MI455X (gfx1250) implementation.
//   BS=2, N=512, C=64, WM=32, CS=128, FD=64, CP=128, E=32768
// Spatial layout for all trunk tensors: [BS][Ch][512][3] -> per-(b,ch) 1536 f32.
// Dominant work done with V_WMMA_F32_16X16X32_BF16 (fp32 accumulate).
// Conv trunk: implicit-im2col GEMM, K staged 128 at a time (4 WMMA / barrier).
// ---------------------------------------------------------------------------

typedef __attribute__((ext_vector_type(16))) __bf16 v16bf;
typedef __attribute__((ext_vector_type(8)))  float  v8f;

#define M3 1536           // 512*3 spatial per batch
#define MTOT 3072         // BS * M3
#define PI_F 3.14159265358979323846f

__device__ __forceinline__ v8f wmma_bf16(v16bf a, v16bf b, v8f c) {
  return __builtin_amdgcn_wmma_f32_16x16x32_bf16(false, a, false, b, (short)0, c,
                                                 false, false);
}

// ========================= small prep kernels ==============================

__global__ void zero_f(float* p, int n) {
  int i = blockIdx.x * 256 + threadIdx.x;
  if (i < n) p[i] = 0.f;
}

// radial + segment_sum over row index
__global__ void radial_agg(const float* __restrict__ pos, const int* __restrict__ ei,
                           float* __restrict__ agg) {
  int e = blockIdx.x * 256 + threadIdx.x;
  if (e >= 32768) return;
  int r = ei[e];
  int c = ei[32768 + e];
  float dx = pos[r * 3 + 0] - pos[c * 3 + 0];
  float dy = pos[r * 3 + 1] - pos[c * 3 + 1];
  float dz = pos[r * 3 + 2] - pos[c * 3 + 2];
  atomicAdd(&agg[r], dx * dx + dy * dy + dz * dz);
}

// NodeEmbedder: pe(sincos CS=128)*charge || at@Waat  -> @nlin -> node[BS*N][3]
__global__ void node_embed(const float* __restrict__ charges, const float* __restrict__ at,
                           const float* __restrict__ Waat, const float* __restrict__ baat,
                           const float* __restrict__ Wn, const float* __restrict__ bn_,
                           float* __restrict__ node) {
  int n = blockIdx.x;           // 0..1023 (b = n/512, i = n%512)
  int t = threadIdx.x;          // 128 threads
  __shared__ float f[256];
  float idx = (float)(n & 511);
  int kf = t & 63;
  float ang = idx * PI_F / powf(2056.0f, 2.0f * (float)kf / 128.0f);
  float pe = (t < 64) ? sinf(ang) : cosf(ang);
  f[t] = pe * charges[n];
  float s = baat[t];
  for (int j = 0; j < 5; ++j) s += at[n * 5 + j] * Waat[j * 128 + t];
  f[128 + t] = s;
  __syncthreads();
  if (t < 3) {
    float o = bn_[t];
    for (int k = 0; k < 256; ++k) o += f[k] * Wn[k * 3 + t];
    node[n * 3 + t] = o;
  }
}

// p_i = [node(3), agg(1)] @ sp(4,64) + b  -> bf16 [BS*N][64]
__global__ void compute_pi(const float* __restrict__ node, const float* __restrict__ agg,
                           const float* __restrict__ Wsp, const float* __restrict__ bsp,
                           __bf16* __restrict__ pib) {
  int id = blockIdx.x * 256 + threadIdx.x;
  if (id >= 1024 * 64) return;
  int n = id >> 6, k = id & 63;
  float v = bsp[k] + node[n * 3 + 0] * Wsp[k] + node[n * 3 + 1] * Wsp[64 + k] +
            node[n * 3 + 2] * Wsp[128 + k] + agg[n] * Wsp[192 + k];
  pib[id] = (__bf16)v;
}

// sincos(d, FD=64) table for d = i-j in [-511,511] -> tab[1023][64] bf16
__global__ void build_sintab(__bf16* __restrict__ tab) {
  int id = blockIdx.x * 256 + threadIdx.x;
  if (id >= 1023 * 64) return;
  int d = id >> 6, k = id & 63;
  float diff = (float)(d - 511);
  int kf = k & 31;
  float ang = diff * PI_F / powf(2056.0f, 2.0f * (float)kf / 64.0f);
  tab[id] = (__bf16)((k < 32) ? sinf(ang) : cosf(ang));
}

// Fold relpos linear into layer-1 weights: Wfold[64][128] = Wr @ W1[128:192,:]
// b1eff[n] = b1[n] + br @ W1[128:192, n]
__global__ void fold_relpos(const float* __restrict__ Wr, const float* __restrict__ br,
                            const float* __restrict__ W1, const float* __restrict__ b1,
                            float* __restrict__ Wfold, float* __restrict__ b1eff) {
  int id = blockIdx.x * 256 + threadIdx.x;
  if (id >= 64 * 128) return;
  int fi = id >> 7, n = id & 127;
  float s = 0.f;
  for (int fo = 0; fo < 64; ++fo) s += Wr[fi * 64 + fo] * W1[(128 + fo) * 128 + n];
  Wfold[id] = s;
  if (fi == 0) {
    float t = b1[n];
    for (int fo = 0; fo < 64; ++fo) t += br[fo] * W1[(128 + fo) * 128 + n];
    b1eff[n] = t;
  }
}

// W1t[128 cols][192 K] bf16 : k<128 from ee1, k>=128 from Wfold
__global__ void pack_w1t(const float* __restrict__ W1, const float* __restrict__ Wfold,
                         __bf16* __restrict__ W1t) {
  int id = blockIdx.x * 256 + threadIdx.x;
  if (id >= 128 * 192) return;
  int n = id / 192, k = id % 192;
  float v = (k < 128) ? W1[k * 128 + n] : Wfold[(k - 128) * 128 + n];
  W1t[id] = (__bf16)v;
}

// Wt[n*K+k] = W[k*N+n]  (fp32 -> bf16 transpose)
__global__ void transpose_bf16(const float* __restrict__ W, __bf16* __restrict__ Wt,
                               int K, int N) {
  int id = blockIdx.x * 256 + threadIdx.x;
  if (id >= K * N) return;
  int n = id / K, k = id % K;
  Wt[id] = (__bf16)W[k * N + n];
}

// conv weight [Cout][Cin*9] fp32 -> bf16 padded [Cout][Kpad] (Kpad % 128 == 0)
__global__ void conv_w_bf16(const float* __restrict__ w, __bf16* __restrict__ wt,
                            int Cout, int K, int Kpad) {
  int id = blockIdx.x * 256 + threadIdx.x;
  if (id >= Cout * Kpad) return;
  int co = id / Kpad, k = id % Kpad;
  wt[id] = (__bf16)((k < K) ? w[co * K + k] : 0.f);
}

// ===================== fused EdgeEmbedder pair-MLP =========================
// grid = BS*N blocks (one per (b, j)), 256 threads = 8 waves.
// Each i-chunk of 16 rows: 16x192 @ 192x128 (relu) @128x128 (relu) @128x128
// -> LayerNorm -> @128x3 -> accumulate over i.  14 v_wmma per chunk per wave;
// weight B-fragments are reloaded from global (W1t/W2t/W3t) in every chunk.
__global__ void __launch_bounds__(256)
edge_mlp(const __bf16* __restrict__ pib, const __bf16* __restrict__ sintab,
         const __bf16* __restrict__ W1t, const float* __restrict__ b1eff,
         const __bf16* __restrict__ W2t, const float* __restrict__ b2,
         const __bf16* __restrict__ W3t, const float* __restrict__ b3,
         const float* __restrict__ lng, const float* __restrict__ lnb,
         const float* __restrict__ Wef, const float* __restrict__ bef,
         float* __restrict__ edge_out) {
  __shared__ __bf16 A[16][200];
  __shared__ __bf16 H[16][136];
  __shared__ float  F[16][132];
  __shared__ float  rowm[16], rowr[16];
  __shared__ float  accv[3];
  int bj = blockIdx.x;
  int b = bj >> 9, j = bj & 511;
  int t = threadIdx.x, lane = t & 31, wave = t >> 5;
  if (t < 3) accv[t] = 0.f;
  const __bf16* pjb = pib + ((size_t)(b * 512 + j) << 6);
  int c0 = wave << 4;
  int col = c0 + (lane & 15);
  int hi = (lane >> 4) << 3;

  for (int i0 = 0; i0 < 512; i0 += 16) {
    __syncthreads();
    // stage A[16][192] = [p_i(i0+r) | p_j | sincos(i0+r-j)]
    for (int e = t; e < 16 * 192; e += 256) {
      int r = e / 192, k = e - r * 192;
      __bf16 v;
      if (k < 64)        v = pib[((size_t)(b * 512 + i0 + r) << 6) + k];
      else if (k < 128)  v = pjb[k - 64];
      else               v = sintab[((size_t)(i0 + r - j + 511) << 6) + (k - 128)];
      A[r][k] = v;
    }
    __syncthreads();

    // ---- layer 1 : K = 192
    v8f acc = {};
#pragma unroll
    for (int kt = 0; kt < 6; ++kt) {
      v16bf a, bb;
      const __bf16* pa = &A[lane & 15][kt * 32 + hi];
      const __bf16* pb = W1t + (size_t)col * 192 + kt * 32 + hi;
#pragma unroll
      for (int q = 0; q < 8; ++q) {
        a[q] = pa[q];  a[q + 8] = pa[q + 16];
        bb[q] = pb[q]; bb[q + 8] = pb[q + 16];
      }
      acc = wmma_bf16(a, bb, acc);
    }
    {
      float bv = b1eff[col];
#pragma unroll
      for (int v = 0; v < 8; ++v) {
        float x = acc[v] + bv;
        H[v + hi][col] = (__bf16)(x > 0.f ? x : 0.f);
      }
    }
    __syncthreads();

    // ---- layer 2 : K = 128
    v8f acc2 = {};
#pragma unroll
    for (int kt = 0; kt < 4; ++kt) {
      v16bf a, bb;
      const __bf16* pa = &H[lane & 15][kt * 32 + hi];
      const __bf16* pb = W2t + (size_t)col * 128 + kt * 32 + hi;
#pragma unroll
      for (int q = 0; q < 8; ++q) {
        a[q] = pa[q];  a[q + 8] = pa[q + 16];
        bb[q] = pb[q]; bb[q + 8] = pb[q + 16];
      }
      acc2 = wmma_bf16(a, bb, acc2);
    }
    __syncthreads();                 // all H reads done
    {
      float bv = b2[col];
#pragma unroll
      for (int v = 0; v < 8; ++v) {
        float x = acc2[v] + bv;
        H[v + hi][col] = (__bf16)(x > 0.f ? x : 0.f);
      }
    }
    __syncthreads();

    // ---- layer 3 : K = 128 -> F (fp32, with bias)
    v8f acc3 = {};
#pragma unroll
    for (int kt = 0; kt < 4; ++kt) {
      v16bf a, bb;
      const __bf16* pa = &H[lane & 15][kt * 32 + hi];
      const __bf16* pb = W3t + (size_t)col * 128 + kt * 32 + hi;
#pragma unroll
      for (int q = 0; q < 8; ++q) {
        a[q] = pa[q];  a[q + 8] = pa[q + 16];
        bb[q] = pb[q]; bb[q + 8] = pb[q + 16];
      }
      acc3 = wmma_bf16(a, bb, acc3);
    }
    {
      float bv = b3[col];
#pragma unroll
      for (int v = 0; v < 8; ++v) F[v + hi][col] = acc3[v] + bv;
    }
    __syncthreads();

    // ---- LayerNorm per row (128 features)
    if (t < 16) {
      float s = 0.f, ss = 0.f;
      for (int k = 0; k < 128; ++k) { float v = F[t][k]; s += v; ss += v * v; }
      float m = s * (1.f / 128.f);
      float var = ss * (1.f / 128.f) - m * m;
      rowm[t] = m;
      rowr[t] = rsqrtf(var + 1e-5f);
    }
    __syncthreads();

    // ---- project to 3 and reduce over the 16 i-rows
    if (t < 48) {
      int r = t / 3, o = t - r * 3;
      float m = rowm[r], rs = rowr[r];
      float s = 0.f;
      for (int k = 0; k < 128; ++k) {
        float v = (F[r][k] - m) * rs * lng[k] + lnb[k];
        s += v * Wef[k * 3 + o];
      }
      atomicAdd(&accv[o], s);
    }
  }
  __syncthreads();
  if (t < 3) edge_out[(size_t)(b * 512 + j) * 3 + t] = accv[t] + 512.f * bef[t];
}

// ==================== implicit-im2col WMMA conv3x3 =========================
// x [BS][Cin][512][3], wt bf16 [Cout][Kpad] (Kpad%128==0, zero padded).
// grid = (192, ceil(Cout/128)), 256 threads = 8 waves (16 N-cols each).
// K staged 128 at a time -> 4 back-to-back WMMA per barrier pair.
__global__ void __launch_bounds__(256)
conv3x3_wmma(const float* __restrict__ x, const __bf16* __restrict__ wt,
             const float* __restrict__ bias, float* __restrict__ y,
             int Cin, int Cout, int Kpad, int dstC, int dstOff) {
  __shared__ __bf16 As[16][132];     // 128 + 4 pad
  int lane = threadIdx.x & 31, wave = threadIdx.x >> 5;
  int mt = blockIdx.x;
  int c0 = blockIdx.y * 128 + wave * 16;
  int K = Cin * 9;
  int col = c0 + (lane & 15);
  int hi = (lane >> 4) << 3;
  bool okc = (col < Cout);
  int colc = okc ? col : 0;
  v8f acc = {};
  int nk = Kpad >> 7;
  for (int kt = 0; kt < nk; ++kt) {
    // stage 16 rows x 128 K of im2col data
    for (int e = threadIdx.x; e < 2048; e += 256) {
      int row = e >> 7, kk = e & 127;
      int m = mt * 16 + row;
      int b = m / M3; int rem = m - b * M3; int h = rem / 3; int w = rem - h * 3;
      int k = (kt << 7) + kk;
      float val = 0.f;
      if (k < K) {
        int c = k / 9; int r9 = k - c * 9;
        int hh = h + r9 / 3 - 1; int ww = w + r9 % 3 - 1;
        if (hh >= 0 && hh < 512 && ww >= 0 && ww < 3)
          val = x[((size_t)(b * Cin + c) * 512 + hh) * 3 + ww];
      }
      As[row][kk] = (__bf16)val;
    }
    __syncthreads();
    const __bf16* pbrow = wt + (size_t)colc * Kpad + (kt << 7) + hi;
#pragma unroll
    for (int s = 0; s < 4; ++s) {
      v16bf a, bb;
      const __bf16* pa = &As[lane & 15][s * 32 + hi];
      const __bf16* pb = pbrow + s * 32;
#pragma unroll
      for (int q = 0; q < 8; ++q) {
        a[q] = pa[q];  a[q + 8] = pa[q + 16];
        __bf16 b0 = pb[q], b1 = pb[q + 16];
        bb[q] = okc ? b0 : (__bf16)0.f;
        bb[q + 8] = okc ? b1 : (__bf16)0.f;
      }
      acc = wmma_bf16(a, bb, acc);
    }
    __syncthreads();
  }
  if (okc) {
    float bv = bias ? bias[col] : 0.f;
#pragma unroll
    for (int v = 0; v < 8; ++v) {
      int m = mt * 16 + v + hi;
      int b = m / M3; int rem = m - b * M3;
      y[(size_t)(b * dstC + dstOff + col) * M3 + rem] = acc[v] + bv;
    }
  }
}

// ========================= BN / glue kernels ===============================

__global__ void bn_stats(const float* __restrict__ y, float* __restrict__ mean,
                         float* __restrict__ rstd, int Cout) {
  int c = blockIdx.x;
  __shared__ float s1[256], s2[256];
  float a = 0.f, q = 0.f;
  for (int i = threadIdx.x; i < MTOT; i += 256) {
    int b = (i >= M3) ? 1 : 0;
    int rem = i - b * M3;
    float v = y[(size_t)(b * Cout + c) * M3 + rem];
    a += v; q += v * v;
  }
  s1[threadIdx.x] = a; s2[threadIdx.x] = q;
  __syncthreads();
  for (int st = 128; st > 0; st >>= 1) {
    if (threadIdx.x < st) {
      s1[threadIdx.x] += s1[threadIdx.x + st];
      s2[threadIdx.x] += s2[threadIdx.x + st];
    }
    __syncthreads();
  }
  if (threadIdx.x == 0) {
    float m = s1[0] * (1.f / (float)MTOT);
    float v = s2[0] * (1.f / (float)MTOT) - m * m;
    mean[c] = m;
    rstd[c] = rsqrtf(v + 1e-5f);
  }
}

__global__ void bn_norm(const float* __restrict__ y, const float* __restrict__ mean,
                        const float* __restrict__ rstd, const float* __restrict__ g,
                        const float* __restrict__ be, float* __restrict__ dst,
                        int Cout, int dstC, int dstOff, int leaky) {
  int idx = blockIdx.x * 256 + threadIdx.x;
  if (idx >= 2 * Cout * M3) return;
  int rem = idx % M3; int tt = idx / M3; int c = tt % Cout; int b = tt / Cout;
  float v = y[idx];
  v = g[c] * (v - mean[c]) * rstd[c] + be[c];
  if (leaky) v = v > 0.f ? v : 0.01f * v;
  dst[(size_t)(b * dstC + dstOff + c) * M3 + rem] = v;
}

__global__ void copy_ch(const float* __restrict__ src, float* __restrict__ dst,
                        int Cs, int dstC, int off) {
  int idx = blockIdx.x * 256 + threadIdx.x;
  if (idx >= 2 * Cs * M3) return;
  int rem = idx % M3; int tt = idx / M3; int c = tt % Cs; int b = tt / Cs;
  dst[(size_t)(b * dstC + off + c) * M3 + rem] = src[idx];
}

__global__ void bcast_msg(const float* __restrict__ msg, float* __restrict__ dst,
                          int dstC, int off) {
  int idx = blockIdx.x * 256 + threadIdx.x;
  if (idx >= 2 * 32 * M3) return;
  int rem = idx % M3; int tt = idx / M3; int m = tt & 31; int b = tt >> 5;
  dst[(size_t)(b * dstC + off + m) * M3 + rem] = msg[b * 32 + m];
}

__global__ void conv_first_k(const float* __restrict__ x, const float* __restrict__ w,
                             const float* __restrict__ bias, float* __restrict__ dst,
                             int dstC) {
  int id = blockIdx.x * 256 + threadIdx.x;
  if (id >= 2 * 64 * M3) return;
  int rem = id % M3; int tt = id / M3; int co = tt & 63; int b = tt >> 6;
  int h = rem / 3, ww = rem - h * 3;
  float s = bias[co];
  for (int ky = 0; ky < 3; ++ky)
    for (int kx = 0; kx < 3; ++kx) {
      int hh = h + ky - 1, wx = ww + kx - 1;
      if (hh >= 0 && hh < 512 && wx >= 0 && wx < 3)
        s += x[b * M3 + hh * 3 + wx] * w[co * 9 + ky * 3 + kx];
    }
  dst[(size_t)(b * dstC + co) * M3 + rem] = s;
}

__global__ void softmax32(const float* __restrict__ x, float* __restrict__ y) {
  int p = blockIdx.x * 256 + threadIdx.x;
  if (p >= MTOT) return;
  int b = (p >= M3) ? 1 : 0;
  int rem = p - b * M3;
  const float* xp = x + (size_t)b * 32 * M3 + rem;
  float mx = -1e30f;
  for (int c = 0; c < 32; ++c) { float v = xp[(size_t)c * M3]; mx = v > mx ? v : mx; }
  float s = 0.f;
  float e[32];
  for (int c = 0; c < 32; ++c) { e[c] = expf(xp[(size_t)c * M3] - mx); s += e[c]; }
  float inv = 1.f / s;
  float* yp = y + (size_t)b * 32 * M3 + rem;
  for (int c = 0; c < 32; ++c) yp[(size_t)c * M3] = e[c] * inv;
}

__global__ void mul_k(const float* __restrict__ a, const float* __restrict__ b,
                      float* __restrict__ o, int n) {
  int i = blockIdx.x * 256 + threadIdx.x;
  if (i < n) o[i] = a[i] * b[i];
}

__global__ void conv_final_k(const float* __restrict__ x, const float* __restrict__ w,
                             const float* __restrict__ bias, const float* __restrict__ pos,
                             float* __restrict__ out) {
  int p = blockIdx.x * 256 + threadIdx.x;
  if (p >= MTOT) return;
  int b = (p >= M3) ? 1 : 0;
  int rem = p - b * M3;
  int h = rem / 3, ww = rem - h * 3;
  float s = bias[0];
  for (int c = 0; c < 32; ++c)
    for (int ky = 0; ky < 3; ++ky)
      for (int kx = 0; kx < 3; ++kx) {
        int hh = h + ky - 1, wx = ww + kx - 1;
        if (hh >= 0 && hh < 512 && wx >= 0 && wx < 3)
          s += x[((size_t)(b * 32 + c) * 512 + hh) * 3 + wx] * w[c * 9 + ky * 3 + kx];
      }
  out[p] = s + pos[p];
}

// ============================ host driver ==================================
// Input order assumption: setup_inputs() dict insertion order, params flattened
// recursively in insertion order ([W,b] lists in order).
extern "C" void kernel_launch(void* const* d_in, const int* in_sizes, int n_in,
                              void* d_out, int out_size, void* d_ws, size_t ws_size,
                              hipStream_t stream) {
  const float* position   = (const float*)d_in[0];
  const float* message    = (const float*)d_in[1];
  const float* charges    = (const float*)d_in[2];
  const float* atom_types = (const float*)d_in[3];
  const int*   edge_index = (const int*)d_in[4];
  auto F = [&](int i) { return (const float*)d_in[i]; };

  // params (insertion order)
  const float *aat_w = F(5), *aat_b = F(6);
  const float *nlin_w = F(7), *nlin_b = F(8);
  const float *sp_w = F(9), *sp_b = F(10);
  const float *rel_w = F(11), *rel_b = F(12);
  const float *ee1_w = F(13), *ee1_b = F(14);
  const float *ee2_w = F(15), *ee2_b = F(16);
  const float *ee3_w = F(17), *ee3_b = F(18);
  const float *lng = F(19), *lnb = F(20);
  const float *ef_w = F(21), *ef_b = F(22);
  const float *first_w = F(23), *first_b = F(24);
  const float *second_w = F(25), *second_b = F(26);
  // dcp blocks: w1,b1,g1,be1,w2,b2,g2,be2 ; cblp blocks: w,b,g,be
  const int DL1 = 27, DL2 = 35, DL3 = 43, DA1 = 51, DA2 = 59, DA3 = 67;
  const int T1 = 75, T2 = 79, FO1 = 83, FO2 = 87;
  const float *final_w = F(91), *final_b = F(92);

  size_t off = 0;
  auto alloc = [&](size_t bytes) -> char* {
    char* p = (char*)d_ws + off;
    off += (bytes + 255) & ~(size_t)255;
    return p;
  };
  auto allocF = [&](size_t n) { return (float*)alloc(n * 4); };
  auto allocB = [&](size_t n) { return (__bf16*)alloc(n * 2); };
  auto G = [](int n) { return (n + 255) / 256; };

  // ---------------- weight prep (bf16, K padded to 128) ----------------
  auto CW = [&](const float* wF, int Cin, int Cout, int& KpadOut) -> __bf16* {
    int K = Cin * 9, Kpad = (K + 127) & ~127;
    __bf16* p = allocB((size_t)Cout * Kpad);
    conv_w_bf16<<<G(Cout * Kpad), 256, 0, stream>>>(wF, p, Cout, K, Kpad);
    KpadOut = Kpad;
    return p;
  };
  int kSecond, kDl1a, kDl1b, kDl2a, kDl2b, kDl3a, kDl3b;
  int kDa1a, kDa1b, kDa2a, kDa2b, kDa3a, kDa3b, kT1, kT2, kFo1, kFo2;
  __bf16* wSecond = CW(second_w, 66, 64, kSecond);
  __bf16* wDl1a = CW(F(DL1 + 0), 96, 256, kDl1a);
  __bf16* wDl1b = CW(F(DL1 + 4), 352, 64, kDl1b);
  __bf16* wDl2a = CW(F(DL2 + 0), 160, 256, kDl2a);
  __bf16* wDl2b = CW(F(DL2 + 4), 416, 64, kDl2b);
  __bf16* wDl3a = CW(F(DL3 + 0), 224, 256, kDl3a);
  __bf16* wDl3b = CW(F(DL3 + 4), 480, 64, kDl3b);
  __bf16* wDa1a = CW(F(DA1 + 0), 64, 512, kDa1a);
  __bf16* wDa1b = CW(F(DA1 + 4), 576, 128, kDa1b);
  __bf16* wDa2a = CW(F(DA2 + 0), 128, 768, kDa2a);
  __bf16* wDa2b = CW(F(DA2 + 4), 896, 192, kDa2b);
  __bf16* wDa3a = CW(F(DA3 + 0), 192, 256, kDa3a);
  __bf16* wDa3b = CW(F(DA3 + 4), 448, 64, kDa3b);
  __bf16* wT1  = CW(F(T1 + 0), 96, 64, kT1);
  __bf16* wT2  = CW(F(T2 + 0), 64, 32, kT2);
  __bf16* wFo1 = CW(F(FO1 + 0), 64, 64, kFo1);
  __bf16* wFo2 = CW(F(FO2 + 0), 64, 32, kFo2);

  // edge weights
  float* Wfold = allocF(64 * 128);
  float* b1eff = allocF(128);
  __bf16* W1t = allocB(128 * 192);
  __bf16* W2t = allocB(128 * 128);
  __bf16* W3t = allocB(128 * 128);
  __bf16* sintab = allocB(1023 * 64);
  fold_relpos<<<G(64 * 128), 256, 0, stream>>>(rel_w, rel_b, ee1_w, ee1_b, Wfold, b1eff);
  pack_w1t<<<G(128 * 192), 256, 0, stream>>>(ee1_w, Wfold, W1t);
  transpose_bf16<<<G(128 * 128), 256, 0, stream>>>(ee2_w, W2t, 128, 128);
  transpose_bf16<<<G(128 * 128), 256, 0, stream>>>(ee3_w, W3t, 128, 128);
  build_sintab<<<G(1023 * 64), 256, 0, stream>>>(sintab);

  // ---------------- activations ----------------
  float* agg = allocF(1024);
  float* node = allocF(3072);
  __bf16* pib = allocB(1024 * 64);
  float* edge = allocF(3072);
  float* cat2 = allocF((size_t)2 * 66 * M3);
  float* femb = allocF((size_t)2 * 64 * M3);
  float* catDl1 = allocF((size_t)2 * 96 * M3);
  float* midDl1 = allocF((size_t)2 * 352 * M3);
  float* f1 = allocF((size_t)2 * 64 * M3);
  float* catDl2 = allocF((size_t)2 * 160 * M3);
  float* midDl2 = allocF((size_t)2 * 416 * M3);
  float* f2 = allocF((size_t)2 * 64 * M3);
  float* catDl3 = allocF((size_t)2 * 224 * M3);
  float* midDl3 = allocF((size_t)2 * 480 * M3);
  float* f3 = allocF((size_t)2 * 64 * M3);
  float* catT = allocF((size_t)2 * 96 * M3);
  float* t1o = allocF((size_t)2 * 64 * M3);
  float* f4 = allocF((size_t)2 * 32 * M3);
  float* midDa1 = allocF((size_t)2 * 576 * M3);
  float* fa1 = allocF((size_t)2 * 128 * M3);
  float* midDa2 = allocF((size_t)2 * 896 * M3);
  float* fa2 = allocF((size_t)2 * 192 * M3);
  float* midDa3 = allocF((size_t)2 * 448 * M3);
  float* fa3 = allocF((size_t)2 * 64 * M3);
  float* fo1o = allocF((size_t)2 * 64 * M3);
  float* fo2o = allocF((size_t)2 * 32 * M3);
  float* mask = allocF((size_t)2 * 32 * M3);
  float* fmul = allocF((size_t)2 * 32 * M3);
  float* ybuf = allocF((size_t)2 * 768 * M3);
  float* meanb = allocF(768);
  float* rstdb = allocF(768);

  // ---------------- embeddings + edge MLP ----------------
  zero_f<<<G(1024), 256, 0, stream>>>(agg, 1024);
  radial_agg<<<G(32768), 256, 0, stream>>>(position, edge_index, agg);
  node_embed<<<1024, 128, 0, stream>>>(charges, atom_types, aat_w, aat_b,
                                       nlin_w, nlin_b, node);
  compute_pi<<<G(1024 * 64), 256, 0, stream>>>(node, agg, sp_w, sp_b, pib);
  edge_mlp<<<1024, 256, 0, stream>>>(pib, sintab, W1t, b1eff, W2t, ee2_b,
                                     W3t, ee3_b, lng, lnb, ef_w, ef_b, edge);

  // ---------------- conv trunk ----------------
  auto CONV = [&](const float* x, __bf16* wB, const float* bias, float* y,
                  int Cin, int Cout, int Kpad, int dstC, int dstOff) {
    conv3x3_wmma<<<dim3(192, (Cout + 127) / 128), 256, 0, stream>>>(
        x, wB, bias, y, Cin, Cout, Kpad, dstC, dstOff);
  };
  auto CBL = [&](const float* x, int Cin, __bf16* wB, int Kpad, int Cout,
                 const float* g, const float* be, float* dst, int dstC, int dstOff) {
    CONV(x, wB, nullptr, ybuf, Cin, Cout, Kpad, Cout, 0);
    bn_stats<<<Cout, 256, 0, stream>>>(ybuf, meanb, rstdb, Cout);
    bn_norm<<<G(2 * Cout * M3), 256, 0, stream>>>(ybuf, meanb, rstdb, g, be,
                                                  dst, Cout, dstC, dstOff, 1);
  };
  auto DCB = [&](const float* xin, int Cin, int Cout, __bf16* w1B, int K1,
                 const float* g1, const float* be1, __bf16* w2B, int K2,
                 const float* g2, const float* be2, float* catMid, float* out) {
    int inner = 4 * Cout;
    CBL(xin, Cin, w1B, K1, inner, g1, be1, catMid, Cin + inner, Cin);
    copy_ch<<<G(2 * Cin * M3), 256, 0, stream>>>(xin, catMid, Cin, Cin + inner, 0);
    CBL(catMid, Cin + inner, w2B, K2, Cout, g2, be2, out, Cout, 0);
  };

  // first + second
  conv_first_k<<<G(2 * 64 * M3), 256, 0, stream>>>(position, first_w, first_b, cat2, 66);
  copy_ch<<<G(2 * 1 * M3), 256, 0, stream>>>(node, cat2, 1, 66, 64);
  copy_ch<<<G(2 * 1 * M3), 256, 0, stream>>>(edge, cat2, 1, 66, 65);
  CONV(cat2, wSecond, second_b, femb, 66, 64, kSecond, 64, 0);

  // f1 = dc([femb,msg], dl1)
  copy_ch<<<G(2 * 64 * M3), 256, 0, stream>>>(femb, catDl1, 64, 96, 0);
  bcast_msg<<<G(2 * 32 * M3), 256, 0, stream>>>(message, catDl1, 96, 64);
  DCB(catDl1, 96, 64, wDl1a, kDl1a, F(DL1 + 2), F(DL1 + 3),
      wDl1b, kDl1b, F(DL1 + 6), F(DL1 + 7), midDl1, f1);

  // f2 = dc([femb,msg,f1], dl2)
  copy_ch<<<G(2 * 64 * M3), 256, 0, stream>>>(femb, catDl2, 64, 160, 0);
  bcast_msg<<<G(2 * 32 * M3), 256, 0, stream>>>(message, catDl2, 160, 64);
  copy_ch<<<G(2 * 64 * M3), 256, 0, stream>>>(f1, catDl2, 64, 160, 96);
  DCB(catDl2, 160, 64, wDl2a, kDl2a, F(DL2 + 2), F(DL2 + 3),
      wDl2b, kDl2b, F(DL2 + 6), F(DL2 + 7), midDl2, f2);

  // f3 = dc([femb,msg,f1,f2], dl3)
  copy_ch<<<G(2 * 64 * M3), 256, 0, stream>>>(femb, catDl3, 64, 224, 0);
  bcast_msg<<<G(2 * 32 * M3), 256, 0, stream>>>(message, catDl3, 224, 64);
  copy_ch<<<G(2 * 64 * M3), 256, 0, stream>>>(f1, catDl3, 64, 224, 96);
  copy_ch<<<G(2 * 64 * M3), 256, 0, stream>>>(f2, catDl3, 64, 224, 160);
  DCB(catDl3, 224, 64, wDl3a, kDl3a, F(DL3 + 2), F(DL3 + 3),
      wDl3b, kDl3b, F(DL3 + 6), F(DL3 + 7), midDl3, f3);

  // f4 = cbl(cbl([f3,msg], t1), t2)
  copy_ch<<<G(2 * 64 * M3), 256, 0, stream>>>(f3, catT, 64, 96, 0);
  bcast_msg<<<G(2 * 32 * M3), 256, 0, stream>>>(message, catT, 96, 64);
  CBL(catT, 96, wT1, kT1, 64, F(T1 + 2), F(T1 + 3), t1o, 64, 0);
  CBL(t1o, 64, wT2, kT2, 32, F(T2 + 2), F(T2 + 3), f4, 32, 0);

  // fa = dc(dc(dc(femb, da1), da2), da3)
  DCB(femb, 64, 128, wDa1a, kDa1a, F(DA1 + 2), F(DA1 + 3),
      wDa1b, kDa1b, F(DA1 + 6), F(DA1 + 7), midDa1, fa1);
  DCB(fa1, 128, 192, wDa2a, kDa2a, F(DA2 + 2), F(DA2 + 3),
      wDa2b, kDa2b, F(DA2 + 6), F(DA2 + 7), midDa2, fa2);
  DCB(fa2, 192, 64, wDa3a, kDa3a, F(DA3 + 2), F(DA3 + 3),
      wDa3b, kDa3b, F(DA3 + 6), F(DA3 + 7), midDa3, fa3);

  // mask = softmax(cbl(cbl(fa, fo1), fo2), axis=ch)
  CBL(fa3, 64, wFo1, kFo1, 64, F(FO1 + 2), F(FO1 + 3), fo1o, 64, 0);
  CBL(fo1o, 64, wFo2, kFo2, 32, F(FO2 + 2), F(FO2 + 3), fo2o, 32, 0);
  softmax32<<<G(MTOT), 256, 0, stream>>>(fo2o, mask);

  // pred = conv(f4*mask, final) + pos
  mul_k<<<G(2 * 32 * M3), 256, 0, stream>>>(f4, mask, fmul, 2 * 32 * M3);
  conv_final_k<<<G(MTOT), 256, 0, stream>>>(fmul, final_w, final_b, position,
                                            (float*)d_out);
}
